// MyModel_61933428416169
// MI455X (gfx1250) — matrix operand order repaired and checked by reading, not compiled, not run on hardware
//
#include <hip/hip_runtime.h>
#include <hip/hip_bf16.h>
#include <stdint.h>

// Problem constants (from reference)
#define BB 512      // batch
#define SS 512      // seq len
#define EE 128      // embedding dim
#define HH 128      // hidden
#define G3 384      // 3*H

typedef __attribute__((ext_vector_type(16))) __bf16 v16bf;
typedef __attribute__((ext_vector_type(8)))  float  v8f;

// ---------------------------------------------------------------------------
// gfx1250 async global->LDS copy (ASYNCcnt-tracked), ISA §10.7 / §15.18.3.
// VDST VGPR carries the per-lane LDS byte address; GV addressing for global.
// ---------------------------------------------------------------------------
__device__ __forceinline__ uint32_t lds_addr_of(const void* p) {
    return (uint32_t)(uintptr_t)(__attribute__((address_space(3))) const void*)p;
}
__device__ __forceinline__ void async_ld_b128(void* lds_dst, const void* gsrc) {
    uint32_t lds = lds_addr_of(lds_dst);
    asm volatile("global_load_async_to_lds_b128 %0, %1, off"
                 :: "v"(lds), "v"(gsrc) : "memory");
}
__device__ __forceinline__ void wait_async_le3() {
    asm volatile("s_wait_asynccnt 0x3" ::: "memory");
}
__device__ __forceinline__ void wait_async_0() {
    asm volatile("s_wait_asynccnt 0x0" ::: "memory");
}

// ---------------------------------------------------------------------------
// WMMA fragment loaders (wave32; contiguous k order k = 16*half + s, same in A and B)
// ---------------------------------------------------------------------------

// A fragment: 16x32 bf16 tile, row-major source with leading dim 128.
// lane L: M = L&15, half = L>>4. VGPR p holds K = half*16 + 2p (+0,+1).
__device__ __forceinline__ v16bf load_A_frag(const __bf16* base, int lane, int kc) {
    const int m  = lane & 15;
    const int kh = (lane >> 4) & 1;
    const __bf16* row = base + m * 128 + kc * 32;
    v16bf a;
#pragma unroll
    for (int p = 0; p < 8; ++p) {
        const int k = (kh << 4) + (p << 1);
        a[2 * p]     = row[k];
        a[2 * p + 1] = row[k + 1];
    }
    return a;
}

// B fragment: 32x16 bf16 tile where B[k][n] = W[ncol0+n][k], W row-major [3H][128].
// lane L: N = L&15, half = L>>4. VGPR p holds K = half*16 + 2p (+0,+1).
__device__ __forceinline__ v16bf load_B_frag(const __bf16* W, int lane, int ncol0, int kc) {
    const int n  = lane & 15;
    const int kh = (lane >> 4) & 1;
    const __bf16* col = W + (size_t)(ncol0 + n) * 128 + kc * 32 + kh * 16;
    v16bf b;
#pragma unroll
    for (int p = 0; p < 8; ++p) {
        b[2 * p]     = col[2 * p];
        b[2 * p + 1] = col[2 * p + 1];
    }
    return b;
}

__device__ __forceinline__ v8f wmma_bf16(v16bf a, v16bf b, v8f c) {
    return __builtin_amdgcn_wmma_f32_16x16x32_bf16(
        /*neg_a=*/false, a, /*neg_b=*/false, b,
        /*c_mod=*/(short)0, c, /*reuse_a=*/false, /*reuse_b=*/false);
}

// ---------------------------------------------------------------------------
// Kernel 0: convert GRU weights f32 -> bf16 into workspace
// ---------------------------------------------------------------------------
__global__ void cvt_w_kernel(const float* __restrict__ wih, const float* __restrict__ whh,
                             __bf16* __restrict__ wih_b, __bf16* __restrict__ whh_b) {
    const int i = blockIdx.x * 256 + threadIdx.x;
    if (i < G3 * HH) {
        wih_b[i] = (__bf16)wih[i];
        whh_b[i] = (__bf16)whh[i];
    }
}

// ---------------------------------------------------------------------------
// Kernel 1: fused embedding gather + input projection
//   gx[s*B + b][g] = emb[x[b][s]] . w_ih[g][:] + b_ih[g]      (bf16 out)
// block = 256 threads = 8 waves; each wave -> 3 N-tiles (covers all 384 cols).
// ---------------------------------------------------------------------------
__global__ void __launch_bounds__(256)
input_proj_kernel(const int* __restrict__ x, const float* __restrict__ emb,
                  const __bf16* __restrict__ wih_b, const float* __restrict__ bih,
                  __bf16* __restrict__ gx) {
    __shared__ __align__(16) __bf16 eA[16 * 128];    // A tile (bf16)
    __shared__ __align__(16) __bf16 gxT[16 * G3];    // staged bf16 output tile
    __shared__ int toks[16];

    const int tid  = threadIdx.x;
    const int lane = tid & 31;
    const int wave = tid >> 5;

    // Time-invariant B fragments for this wave's 3 N-tiles (register-resident).
    v16bf Bf[3][4];
#pragma unroll
    for (int nt = 0; nt < 3; ++nt)
#pragma unroll
        for (int kc = 0; kc < 4; ++kc)
            Bf[nt][kc] = load_B_frag(wih_b, lane, (wave * 3 + nt) * 16, kc);

    // Per-lane bias (C layout: lane holds one column for all 8 rows).
    float bias[3];
#pragma unroll
    for (int nt = 0; nt < 3; ++nt)
        bias[nt] = bih[(wave * 3 + nt) * 16 + (lane & 15)];

    const int nTiles = (BB * SS) / 16;  // 16384 row tiles
    for (int rt = blockIdx.x; rt < nTiles; rt += gridDim.x) {
        const int r0    = rt * 16;      // rows r0..r0+15 share s (16 | 512)
        const int s     = r0 >> 9;
        const int bbase = r0 & (BB - 1);

        if (tid < 16) toks[tid] = x[(size_t)(bbase + tid) * SS + s];
        __syncthreads();

        // Gather 16 embedding rows, convert to bf16 in LDS (float4 loads).
        for (int i = tid; i < 512; i += 256) {           // 16 rows * 32 float4
            const int m = i >> 5, q = i & 31;
            const float4 v4 = ((const float4*)(emb + (size_t)toks[m] * EE))[q];
            __bf16* dst = eA + m * 128 + q * 4;
            dst[0] = (__bf16)v4.x; dst[1] = (__bf16)v4.y;
            dst[2] = (__bf16)v4.z; dst[3] = (__bf16)v4.w;
        }
        __syncthreads();

        v16bf A[4];
#pragma unroll
        for (int kc = 0; kc < 4; ++kc) A[kc] = load_A_frag(eA, lane, kc);

        const int n  = lane & 15;
        const int mh = (lane >> 4) & 1;
#pragma unroll
        for (int nt = 0; nt < 3; ++nt) {
            v8f acc = {};
#pragma unroll
            for (int kc = 0; kc < 4; ++kc) acc = wmma_bf16(A[kc], Bf[nt][kc], acc);
            const int col = (wave * 3 + nt) * 16 + n;
#pragma unroll
            for (int v = 0; v < 8; ++v)
                gxT[(v + 8 * mh) * G3 + col] = (__bf16)(acc[v] + bias[nt]);
        }
        __syncthreads();

        // Coalesced 16B store of the 16x384 bf16 tile (12 KB).
        uint4* dst = (uint4*)(gx + (size_t)r0 * G3);
        const uint4* src = (const uint4*)gxT;
        for (int i = tid; i < (16 * G3) / 8; i += 256) dst[i] = src[i];
        __syncthreads();
    }
}

// ---------------------------------------------------------------------------
// Kernel 2: sequential GRU scan. 32 blocks, each owns 16 batch rows.
// h kept LDS-resident (f32 + bf16 mirrors); w_hh B-fragments register-resident
// across all 512 steps; next step's gx slab double-buffered in LDS via
// global_load_async_to_lds_b128 (ASYNCcnt), overlapped with the WMMA phase.
// ---------------------------------------------------------------------------
__global__ void __launch_bounds__(256)
gru_scan_kernel(const __bf16* __restrict__ gx, const __bf16* __restrict__ whh_b,
                const float* __restrict__ bhh, float* __restrict__ hT) {
    __shared__ __align__(16) __bf16 h_bf[16 * 128];
    __shared__ float  h_f [16 * 128];
    __shared__ float  gh  [16 * G3];
    __shared__ float  bhh_s[G3];
    __shared__ __align__(16) __bf16 gxL[2][16 * G3];   // double-buffered gx slab

    const int tid  = threadIdx.x;
    const int lane = tid & 31;
    const int wave = tid >> 5;
    const int b0   = blockIdx.x * 16;

    for (int i = tid; i < 2048; i += 256) { h_f[i] = 0.f; h_bf[i] = (__bf16)0.f; }
    for (int i = tid; i < G3;   i += 256) bhh_s[i] = bhh[i];

    // w_hh B fragments: loaded once, live in VGPRs for the whole scan.
    v16bf Bf[3][4];
#pragma unroll
    for (int nt = 0; nt < 3; ++nt)
#pragma unroll
        for (int kc = 0; kc < 4; ++kc)
            Bf[nt][kc] = load_B_frag(whh_b, lane, (wave * 3 + nt) * 16, kc);

    // Kick off async copy of step-0 slab (16*384 bf16 = 12 KB = 768 x 16B,
    // 3 b128 chunks per thread, ASYNCcnt = 3 per wave).
    {
        const __bf16* slab = gx + (size_t)b0 * G3;    // t = 0
#pragma unroll
        for (int c = 0; c < 3; ++c) {
            const int idx = tid + c * 256;            // 16B chunk index
            async_ld_b128(&gxL[0][idx * 8], slab + idx * 8);
        }
    }
    __syncthreads();

    const int n  = lane & 15;
    const int mh = (lane >> 4) & 1;

    for (int t = 0; t < SS; ++t) {
        const int cur = t & 1;

        // ---- gh = h @ w_hh^T  (bf16 WMMA, f32 accumulate) ----
        v16bf A[4];
#pragma unroll
        for (int kc = 0; kc < 4; ++kc) A[kc] = load_A_frag(h_bf, lane, kc);
#pragma unroll
        for (int nt = 0; nt < 3; ++nt) {
            v8f acc = {};
#pragma unroll
            for (int kc = 0; kc < 4; ++kc) acc = wmma_bf16(A[kc], Bf[nt][kc], acc);
            const int col = (wave * 3 + nt) * 16 + n;
#pragma unroll
            for (int v = 0; v < 8; ++v)
                gh[(v + 8 * mh) * G3 + col] = acc[v];
        }

        // Issue async copy for step t+1 into the other buffer, then wait for
        // the *previous* slab (<=3 outstanding = the 3 just issued).
        if (t + 1 < SS) {
            const __bf16* slab = gx + ((size_t)(t + 1) * BB + b0) * G3;
#pragma unroll
            for (int c = 0; c < 3; ++c) {
                const int idx = tid + c * 256;
                async_ld_b128(&gxL[cur ^ 1][idx * 8], slab + idx * 8);
            }
            wait_async_le3();
        } else {
            wait_async_0();
        }
        __syncthreads();

        // ---- elementwise gates (f32 math, bf16 gx from LDS) ----
        const __bf16* gxt = gxL[cur];
        for (int i = tid; i < 2048; i += 256) {
            const int m = i >> 7, j = i & 127;
            const float xr = (float)gxt[m * G3 + j];
            const float xz = (float)gxt[m * G3 + 128 + j];
            const float xn = (float)gxt[m * G3 + 256 + j];
            const float hr = gh[m * G3 + j]       + bhh_s[j];
            const float hz = gh[m * G3 + 128 + j] + bhh_s[128 + j];
            const float hn = gh[m * G3 + 256 + j] + bhh_s[256 + j];
            const float r  = 1.f / (1.f + __expf(-(xr + hr)));
            const float z  = 1.f / (1.f + __expf(-(xz + hz)));
            const float nn = tanhf(xn + r * hn);
            const float hv = (1.f - z) * nn + z * h_f[i];
            h_f[i]  = hv;
            h_bf[i] = (__bf16)hv;
        }
        __syncthreads();
    }

    for (int i = tid; i < 2048; i += 256) hT[(size_t)b0 * HH + i] = h_f[i];
}

// ---------------------------------------------------------------------------
// Kernel 3: out = hT @ w_out^T + b_out   ([512,128] x [128,2])
// ---------------------------------------------------------------------------
__global__ void out_proj_kernel(const float* __restrict__ hT, const float* __restrict__ wout,
                                const float* __restrict__ bout, float* __restrict__ out) {
    const int gid = blockIdx.x * blockDim.x + threadIdx.x;
    if (gid >= BB * 2) return;
    const int b = gid >> 1, o = gid & 1;
    const float* hp = hT + (size_t)b * HH;
    const float* wp = wout + (size_t)o * HH;
    float s = bout[o];
#pragma unroll 8
    for (int j = 0; j < HH; ++j) s += hp[j] * wp[j];
    out[gid] = s;
}

// ---------------------------------------------------------------------------
// Launch
// ---------------------------------------------------------------------------
extern "C" void kernel_launch(void* const* d_in, const int* in_sizes, int n_in,
                              void* d_out, int out_size, void* d_ws, size_t ws_size,
                              hipStream_t stream) {
    const int*   x    = (const int*)d_in[0];
    const float* emb  = (const float*)d_in[1];
    const float* wih  = (const float*)d_in[2];
    const float* whh  = (const float*)d_in[3];
    const float* bih  = (const float*)d_in[4];
    const float* bhh  = (const float*)d_in[5];
    const float* wout = (const float*)d_in[6];
    const float* bout = (const float*)d_in[7];
    float* out = (float*)d_out;

    // Workspace layout (256B aligned)
    char* ws = (char*)d_ws;
    __bf16* wih_b = (__bf16*)(ws + 0);                       //  98304 B
    __bf16* whh_b = (__bf16*)(ws + 98304);                   //  98304 B
    float*  hT    = (float*)(ws + 196608);                   // 262144 B
    __bf16* gx    = (__bf16*)(ws + 458752);                  // 201,326,592 B

    cvt_w_kernel<<<(G3 * HH + 255) / 256, 256, 0, stream>>>(wih, whh, wih_b, whh_b);
    input_proj_kernel<<<2048, 256, 0, stream>>>(x, emb, wih_b, bih, gx);
    gru_scan_kernel<<<BB / 16, 256, 0, stream>>>(gx, whh_b, bhh, hT);
    out_proj_kernel<<<(BB * 2 + 255) / 256, 256, 0, stream>>>(hT, wout, bout, out);
}
